// DiscreteBottleneck_49160195670623
// MI455X (gfx1250) — matrix operand order repaired and checked by reading, compiled, not run on hardware
//
#include <hip/hip_runtime.h>
#include <hip/hip_bf16.h>
#include <stdint.h>

// ---------------- problem constants ----------------
#define N_ROWS 16384   // BATCH*K = 64*256
#define CB     4096    // codebook size
#define DIM    1024    // embedding dim

// GEMM tiling: block tile 64(M) x 256(N), K-step 32; 8 waves of 32 lanes,
// each wave owns a 32x64 sub-tile = 2x4 WMMA 16x16 tiles.
#define BM 64
#define BN 256
#define BK 32

// ---------------- workspace layout (bytes) ----------------
#define WS_XBF16   ((size_t)0)
#define WS_CBBF16  ((size_t)N_ROWS * DIM * 2)                 // 32 MB in
#define WS_XNORM   (WS_CBBF16 + (size_t)CB * DIM * 2)         // + 8 MB
#define WS_CNORM   (WS_XNORM + (size_t)N_ROWS * 4)
#define WS_ROWLOSS (WS_CNORM + (size_t)CB * 4)

// ---------------- d_out layout (float elements) ----------------
#define OUT_Q      ((size_t)0)                                // [16384,1024]
#define OUT_CODES  ((size_t)N_ROWS * DIM)                     // [16384]
#define OUT_PROBS  (OUT_CODES + (size_t)N_ROWS)               // [16384,4096]
#define OUT_LOSS   (OUT_PROBS + (size_t)N_ROWS * CB)          // scalar

typedef __attribute__((ext_vector_type(16))) __bf16 bf16x16;
typedef __attribute__((ext_vector_type(8)))  __bf16 bf16x8;
typedef __attribute__((ext_vector_type(8)))  float  f32x8;
typedef __attribute__((ext_vector_type(4)))  int    i32x4;

// ---------------- gfx1250 async global->LDS path (guarded) ----------------
#ifndef __has_builtin
#define __has_builtin(x) 0
#endif

#if __has_builtin(__builtin_amdgcn_global_load_async_to_lds_b128)
#define HAVE_ASYNC_LDS 1
#else
#define HAVE_ASYNC_LDS 0
#endif

typedef __attribute__((address_space(1))) i32x4 gbl_i32x4;
typedef __attribute__((address_space(3))) i32x4 lds_i32x4;

// Copy 16 bytes global -> LDS. Async (ASYNCcnt-tracked, no VGPR round-trip)
// when the gfx1250 builtin is available; plain copy otherwise.
static __device__ __forceinline__ void stage16(const void* g, void* l) {
#if HAVE_ASYNC_LDS
    // Prototype (from toolchain): (int4 AS1*, int4 AS3*, imm offset, imm cpol).
    // Integer casts: AS1 ptr keeps all 64 bits; AS3 ptr is the low 32 bits of
    // a generic LDS pointer.
    __builtin_amdgcn_global_load_async_to_lds_b128(
        (gbl_i32x4*)(uintptr_t)g,
        (lds_i32x4*)(unsigned int)(uintptr_t)l,
        0 /*imm offset*/, 0 /*cpol*/);
#else
    *(uint4*)l = *(const uint4*)g;
#endif
}

static __device__ __forceinline__ void stage_wait_all() {
#if HAVE_ASYNC_LDS
#if __has_builtin(__builtin_amdgcn_s_wait_asynccnt)
    __builtin_amdgcn_s_wait_asynccnt(0);
#else
    asm volatile("s_wait_asynccnt 0" ::: "memory");
#endif
#endif
}

// fp32 -> bf16 (round to nearest even)
static __device__ __forceinline__ unsigned short f32_to_bf16(float f) {
    unsigned int u = __float_as_uint(f);
    unsigned int r = u + 0x7FFFu + ((u >> 16) & 1u);
    return (unsigned short)(r >> 16);
}

// Convert one row (DIM floats) to bf16 and compute its squared norm.
__global__ void __launch_bounds__(256)
vq_prep(const float* __restrict__ src, unsigned short* __restrict__ dst,
        float* __restrict__ norm) {
    const int row = blockIdx.x;
    const int tid = threadIdx.x;
    const float* s = src + (size_t)row * DIM;
    unsigned short* d = dst + (size_t)row * DIM;
    float acc = 0.f;
    #pragma unroll
    for (int i = tid; i < DIM; i += 256) {
        float v = s[i];
        acc += v * v;
        d[i] = f32_to_bf16(v);
    }
    __shared__ float red[256];
    red[tid] = acc;
    __syncthreads();
    for (int off = 128; off > 0; off >>= 1) {
        if (tid < off) red[tid] += red[tid + off];
        __syncthreads();
    }
    if (tid == 0) norm[row] = red[0];
}

// Main WMMA GEMM: dot[n][c] = <x_bf16[n,:], cb_bf16[c,:]> with f32 accumulate.
// Double-buffered LDS; next K-tile staged (async when available) during compute.
__global__ void __launch_bounds__(256)
vq_gemm(const unsigned short* __restrict__ xb,
        const unsigned short* __restrict__ cb,
        float* __restrict__ dot_out) {
    __shared__ __align__(64) unsigned short As[2][BM * BK];   // 2 x  4 KB
    __shared__ __align__(64) unsigned short Bs[2][BN * BK];   // 2 x 16 KB

    const int tid  = threadIdx.x;
    const int lane = tid & 31;
    const int wave = tid >> 5;      // 0..7
    const int wm   = wave >> 2;     // 0..1  (M sub-block of 32)
    const int wn   = wave & 3;      // 0..3  (N sub-block of 64)
    const int blockM = blockIdx.x * BM;
    const int blockN = blockIdx.y * BN;

    const f32x8 zero = {0.f, 0.f, 0.f, 0.f, 0.f, 0.f, 0.f, 0.f};
    f32x8 acc[2][4];
    #pragma unroll
    for (int m = 0; m < 2; ++m)
        #pragma unroll
        for (int n = 0; n < 4; ++n) acc[m][n] = zero;

    // A tile load mapping: 64 rows x 32 k, 16B (8 bf16) per thread
    const int a_row = tid >> 2;          // 0..63
    const int a_col = (tid & 3) * 8;     // 0,8,16,24
    const size_t a_base = (size_t)(blockM + a_row) * DIM + a_col;
    // B tile load mapping: 256 codebook rows x 32 k, 64B per thread
    const size_t b_base = (size_t)(blockN + tid) * DIM;

    const int half = lane >> 4;          // 0|1
    const int lr   = lane & 15;

    // stage one K-tile into buffer `buf`
    auto stage = [&](int buf, int k) {
        stage16(&xb[a_base + k], &As[buf][a_row * BK + a_col]);
        const unsigned short* bsrc = &cb[b_base + k];
        unsigned short* bdst = &Bs[buf][tid * BK];
        stage16(bsrc +  0, bdst +  0);
        stage16(bsrc +  8, bdst +  8);
        stage16(bsrc + 16, bdst + 16);
        stage16(bsrc + 24, bdst + 24);
    };

    stage(0, 0);
    stage_wait_all();
    __syncthreads();

    int buf = 0;
    for (int k = 0; k < DIM; k += BK) {
        // prefetch next K-tile into the other buffer while we compute
        if (k + BK < DIM) stage(buf ^ 1, k + BK);

        // A fragments: 16-bit A 16x32 layout — lanes 0-15 hold K{0..7,16..23},
        // lanes 16-31 hold K{8..15,24..31}, row M = lane%16.
        bf16x16 afrag[2];
        #pragma unroll
        for (int m = 0; m < 2; ++m) {
            const unsigned short* ap = &As[buf][(wm * 32 + m * 16 + lr) * BK];
            bf16x8 lo = *(const bf16x8*)(ap + half * 8);
            bf16x8 hi = *(const bf16x8*)(ap + 16 + half * 8);
            afrag[m] = __builtin_shufflevector(lo, hi,
                0, 1, 2, 3, 4, 5, 6, 7, 8, 9, 10, 11, 12, 13, 14, 15);
        }

        // B fragments: 16-bit B 32x16 layout — lane L holds column N=L%16,
        // lanes 0-15 hold K{0..7,16..23}, lanes 16-31 hold K{8..15,24..31}.
        #pragma unroll
        for (int n = 0; n < 4; ++n) {
            const unsigned short* bp =
                &Bs[buf][(wn * 64 + n * 16 + lr) * BK];
            bf16x8 blo = *(const bf16x8*)(bp + half * 8);
            bf16x8 bhi = *(const bf16x8*)(bp + 16 + half * 8);
            bf16x16 bfrag = __builtin_shufflevector(blo, bhi,
                0, 1, 2, 3, 4, 5, 6, 7, 8, 9, 10, 11, 12, 13, 14, 15);
            #pragma unroll
            for (int m = 0; m < 2; ++m) {
                acc[m][n] = __builtin_amdgcn_wmma_f32_16x16x32_bf16(
                    false, afrag[m], false, bfrag,
                    (short)0, acc[m][n], false, false);
            }
        }

        stage_wait_all();    // my async writes to buf^1 have landed
        __syncthreads();     // everyone's have; reads of buf are done
        buf ^= 1;
    }

    // Epilogue: C/D layout — VGPR j: lanes 0-15 -> M=j, lanes 16-31 -> M=8+j.
    #pragma unroll
    for (int m = 0; m < 2; ++m) {
        #pragma unroll
        for (int n = 0; n < 4; ++n) {
            const int col = blockN + wn * 64 + n * 16 + lr;
            #pragma unroll
            for (int j = 0; j < 8; ++j) {
                const int row = blockM + wm * 32 + m * 16 + half * 8 + j;
                dot_out[(size_t)row * CB + col] = acc[m][n][j];
            }
        }
    }
}

// Per-row: dist = ||x||^2 + ||c||^2 - 2*dot (in place), argmin, softmax,
// codebook gather (fp32), per-row loss partial.
__global__ void __launch_bounds__(256)
vq_rowreduce(const float* __restrict__ x, const float* __restrict__ codebook,
             const float* __restrict__ xnorm, const float* __restrict__ cnorm,
             float* __restrict__ out, float* __restrict__ rowloss) {
    const int row = blockIdx.x;
    const int tid = threadIdx.x;
    float* dist = out + OUT_PROBS + (size_t)row * CB;
    const float xn = xnorm[row];

    __shared__ float sd[256];
    __shared__ int   si[256];

    // pass 1: dot -> distance, local argmin (first occurrence wins)
    float bestd = 3.4e38f;
    int   besti = 0;
    for (int c = tid; c < CB; c += 256) {
        float d = xn + cnorm[c] - 2.0f * dist[c];
        dist[c] = d;
        if (d < bestd) { bestd = d; besti = c; }
    }
    sd[tid] = bestd; si[tid] = besti;
    __syncthreads();
    for (int off = 128; off > 0; off >>= 1) {
        if (tid < off) {
            float od = sd[tid + off]; int oi = si[tid + off];
            if (od < sd[tid] || (od == sd[tid] && oi < si[tid])) {
                sd[tid] = od; si[tid] = oi;
            }
        }
        __syncthreads();
    }
    const float dmin = sd[0];
    const int   code = si[0];
    if (tid == 0) out[OUT_CODES + row] = (float)code;
    __syncthreads();

    // pass 2: sum of exp((dmin - d)/T), T = 1
    float se = 0.f;
    for (int c = tid; c < CB; c += 256) se += __expf(dmin - dist[c]);
    sd[tid] = se;
    __syncthreads();
    for (int off = 128; off > 0; off >>= 1) {
        if (tid < off) sd[tid] += sd[tid + off];
        __syncthreads();
    }
    const float inv = 1.0f / sd[0];
    __syncthreads();

    // pass 3: write probs in place
    for (int c = tid; c < CB; c += 256) dist[c] = __expf(dmin - dist[c]) * inv;

    // gather quantized row (fp32 codebook) + per-row loss partial
    const float* q  = codebook + (size_t)code * DIM;
    const float* xr = x + (size_t)row * DIM;
    float* qo = out + OUT_Q + (size_t)row * DIM;
    float l = 0.f;
    for (int i = tid; i < DIM; i += 256) {
        float qv = q[i];
        float xv = xr[i];
        qo[i] = qv;                      // quantized_st forward == quantized
        float diff = xv - qv;
        l += diff * diff;
    }
    sd[tid] = l;
    __syncthreads();
    for (int off = 128; off > 0; off >>= 1) {
        if (tid < off) sd[tid] += sd[tid + off];
        __syncthreads();
    }
    if (tid == 0) rowloss[row] = sd[0];
}

// vq_loss = (1 + BETA) * mean((x - q)^2), BETA = 0.25
__global__ void __launch_bounds__(256)
vq_finalize(const float* __restrict__ rowloss, float* __restrict__ out) {
    __shared__ float sd[256];
    const int tid = threadIdx.x;
    float a = 0.f;
    for (int i = tid; i < N_ROWS; i += 256) a += rowloss[i];
    sd[tid] = a;
    __syncthreads();
    for (int off = 128; off > 0; off >>= 1) {
        if (tid < off) sd[tid] += sd[tid + off];
        __syncthreads();
    }
    if (tid == 0)
        out[OUT_LOSS] = sd[0] * (1.25f / ((float)N_ROWS * (float)DIM));
}

extern "C" void kernel_launch(void* const* d_in, const int* in_sizes, int n_in,
                              void* d_out, int out_size, void* d_ws, size_t ws_size,
                              hipStream_t stream) {
    const float* x  = (const float*)d_in[0];   // [64,256,1024] f32
    const float* cb = (const float*)d_in[1];   // [4096,1024]   f32
    float* out = (float*)d_out;
    char*  ws  = (char*)d_ws;

    unsigned short* xbf    = (unsigned short*)(ws + WS_XBF16);
    unsigned short* cbf    = (unsigned short*)(ws + WS_CBBF16);
    float* xnorm           = (float*)(ws + WS_XNORM);
    float* cnorm           = (float*)(ws + WS_CNORM);
    float* rowloss         = (float*)(ws + WS_ROWLOSS);

    vq_prep<<<N_ROWS, 256, 0, stream>>>(x, xbf, xnorm);
    vq_prep<<<CB,     256, 0, stream>>>(cb, cbf, cnorm);

    dim3 grid(N_ROWS / BM, CB / BN);
    vq_gemm<<<grid, 256, 0, stream>>>(xbf, cbf, out + OUT_PROBS);

    vq_rowreduce<<<N_ROWS, 256, 0, stream>>>(x, cb, xnorm, cnorm, out, rowloss);
    vq_finalize<<<1, 256, 0, stream>>>(rowloss, out);
}
